// AttModel_18854906429622
// MI455X (gfx1250) — matrix operand order repaired and checked by reading, not compiled, not run on hardware
//
#include <hip/hip_runtime.h>
#include <math.h>

typedef unsigned short bhalf;   // bf16 storage
typedef unsigned int   u32t;

typedef __attribute__((ext_vector_type(16))) __bf16 v16bf;
typedef __attribute__((ext_vector_type(8)))  float  v8f;
typedef __attribute__((ext_vector_type(8)))  u32t   v8u;
typedef __attribute__((ext_vector_type(4)))  u32t   v4u;

constexpr int Bn  = 512;   // batch
constexpr int Tt  = 50;    // input_n
constexpr int Ff  = 48;    // IN_F / spatial nodes
constexpr int Dm  = 512;   // D_MODEL
constexpr int Dct = 10;    // DCT_N
constexpr int Vl  = 35;    // KS + output_n
constexpr int Vn  = 16;    // value windows

constexpr int RM_DENSE = 0, RM_CONV = 1, RM_ATT = 2;

static __device__ __forceinline__ bhalf f2bf(float f){
  union{float f;u32t u;}x; x.f=f;
  u32t r = x.u + 0x7FFFu + ((x.u>>16)&1u);
  return (bhalf)(r>>16);
}
static __device__ __forceinline__ float bf2f(bhalf h){
  union{u32t u;float f;}x; x.u = ((u32t)h)<<16; return x.f;
}

// ------------------------------------------------------------------
// WMMA GEMM, bf16 x bf16 -> f32 accum, all strides/divisors compile-time.
//  RM_DENSE: A row = m,                    col = k
//  RM_CONV : A row = (m/TO)*TI+m%TO+k/KIN, col = k%KIN   (im2col folded)
//  RM_ATT  : A row = m%TO, B batched: brow = (mTile/TO)*Dm + n
// B stored N-major: elem(k,n) at Bm[brow*BSTR + k].
// No bounds checks: every operand zero-padded to its padded K stride.
// Per-lane fragments load as contiguous 16B runs (ISA 7.12.2 layouts):
//   A: k in [k0+8h, +8) and [k0+16+8h, +8)   -> 2 x b128
//   B: k in [k0+8h, +8) and [k0+16+8h, +8)   -> 2 x b128
// MODE 0: relu -> bf16 row-major (stride Dm)
// MODE 1: bf16 transposed per-batch: Obf[(m/ONP*Dm + n)*64 + m%ONP] (16B store)
// MODE 2: (acc+bias)*inv*gamma+beta, tanh, +resid?, -> bf16 row-major
// ------------------------------------------------------------------
template<int RM, int TI, int TO, int KIN, int ASTR, int BSTR, int MODE, int ONP>
__global__ __launch_bounds__(256) void k_wmma_gemm(
    const bhalf* __restrict__ A, const bhalf* __restrict__ Bm,
    int KK, bhalf* __restrict__ Obf, int Nreal,
    const float* __restrict__ bias, const float* __restrict__ gamma,
    const float* __restrict__ beta, const bhalf* __restrict__ resid)
{
  const int lane  = threadIdx.x & 31;
  const int wv    = threadIdx.x >> 5;
  const int mBase = blockIdx.y * 64  + (wv >> 2) * 32;
  const int nBase = blockIdx.x * 256 + (wv & 3) * 64;
  const int l15   = lane & 15;
  const int half  = lane >> 4;

  v8f acc[2][4];
#pragma unroll
  for (int i=0;i<2;i++)
#pragma unroll
    for (int j=0;j<4;j++)
#pragma unroll
      for (int r=0;r<8;r++) acc[i][j][r]=0.f;

  int arow[2], bg[2];
  const bhalf* aBase[2];
#pragma unroll
  for (int mt=0; mt<2; mt++) {
    const int m = mBase + mt*16 + l15;
    if      (RM == RM_DENSE) arow[mt] = m;
    else if (RM == RM_ATT)   arow[mt] = m % TO;
    else                     arow[mt] = (m/TO)*TI + (m%TO);
    aBase[mt] = A + arow[mt]*ASTR + 8*half;    // + k0 (+16) per step; conv recomputes
    bg[mt] = (RM == RM_ATT) ? (mBase + mt*16) / TO : 0;
  }
  const bhalf* bBase[4];
#pragma unroll
  for (int nt=0; nt<4; nt++)
    bBase[nt] = Bm + (nBase + nt*16 + l15)*BSTR + 8*half;   // non-ATT base

  for (int k0=0; k0<KK; k0+=32) {
    v16bf aF[2];
#pragma unroll
    for (int mt=0; mt<2; mt++) {
      v4u lo, hi;
      if (RM == RM_CONV) {
        const int kA = k0 + 8*half;            // low 16B run
        const int jA = kA / KIN;
        lo = *(const v4u*)(A + (arow[mt]+jA)*ASTR + (kA - jA*KIN));
        const int kB = kA + 16;                // high 16B run
        const int jB = kB / KIN;
        hi = *(const v4u*)(A + (arow[mt]+jB)*ASTR + (kB - jB*KIN));
      } else {
        lo = *(const v4u*)(aBase[mt] + k0);
        hi = *(const v4u*)(aBase[mt] + k0 + 16);
      }
      v8u ar;
#pragma unroll
      for (int r=0;r<4;r++) { ar[r] = lo[r]; ar[r+4] = hi[r]; }
      aF[mt] = __builtin_bit_cast(v16bf, ar);
    }
    v16bf bF[4];
#pragma unroll
    for (int mt=0; mt<2; mt++) {
      if (mt == 0 || (RM == RM_ATT && bg[1] != bg[0])) {   // folds when !ATT
#pragma unroll
        for (int nt=0; nt<4; nt++) {
          const bhalf* bp = (RM == RM_ATT)
              ? Bm + (bg[mt]*Dm + nBase + nt*16 + l15)*BSTR + 8*half + k0
              : bBase[nt] + k0;
          v4u blo = *(const v4u*)bp, bhi = *(const v4u*)(bp + 16);
          v8u br;
#pragma unroll
          for (int r=0;r<4;r++) { br[r] = blo[r]; br[r+4] = bhi[r]; }
          bF[nt] = __builtin_bit_cast(v16bf, br);   // 2 x 16B runs
        }
      }
#pragma unroll
      for (int nt=0; nt<4; nt++)
        acc[mt][nt] = __builtin_amdgcn_wmma_f32_16x16x32_bf16(
            false, aF[mt], false, bF[nt], (short)0, acc[mt][nt], false, false);
    }
  }

  const float binv = 0.9999950000374997f;   // 1/sqrt(1 + 1e-5)
#pragma unroll
  for (int mt=0; mt<2; mt++) {
#pragma unroll
    for (int nt=0; nt<4; nt++) {
      const int n = nBase + nt*16 + l15;
      if (MODE == 1) {
        const int m0 = mBase + mt*16 + 8*half;     // 8 consecutive m (ISA C layout)
        const int batch = m0 / ONP, ml0 = m0 % ONP;
        v4u pk;
#pragma unroll
        for (int r=0;r<4;r++)
          pk[r] = (u32t)f2bf(acc[mt][nt][2*r]) | ((u32t)f2bf(acc[mt][nt][2*r+1]) << 16);
        *(v4u*)(Obf + (batch*Dm + n)*64 + ml0) = pk;
      } else {
#pragma unroll
        for (int r=0;r<8;r++) {
          const int m = mBase + mt*16 + r + 8*half;
          float v = acc[mt][nt][r];
          if (MODE == 0) {
            v = v > 0.f ? v : 0.f;
            Obf[m*Dm + n] = f2bf(v);
          } else {
            int node = m % TO; if (node >= Nreal) node = Nreal - 1;
            float y = (v + bias[n]) * binv * gamma[node*Dm + n] + beta[node*Dm + n];
            y = tanhf(y);
            if (resid) y += bf2f(resid[m*Dm + n]);
            Obf[m*Dm + n] = f2bf(y);
          }
        }
      }
    }
  }
}

// ------------------------- glue kernels (VALU) -------------------------
__global__ void k_fill0(u32t* __restrict__ p, int n){
  int i = blockIdx.x*blockDim.x + threadIdx.x;
  if (i < n) p[i] = 0u;
}

__global__ void k_build_dct(float* __restrict__ d){
  int i = blockIdx.x*blockDim.x + threadIdx.x;
  if (i >= Dct*Vl) return;
  int k = i / Vl, t = i % Vl;
  float w = (k==0) ? sqrtf(1.0f/Vl) : sqrtf(2.0f/Vl);
  d[i] = w * cosf(3.14159265358979323846f * (t + 0.5f) * (float)k / (float)Vl);
}

__global__ void k_conv_wT(const float* __restrict__ w, bhalf* __restrict__ wT, int I, int J){
  int idx = blockIdx.x*blockDim.x + threadIdx.x;
  if (idx >= Dm*I*J) return;
  int j = idx % J, i = (idx / J) % I, o = idx / (I*J);
  wT[o*(I*J) + j*I + i] = f2bf(w[idx]);     // w is (O,I,J); k = j*I + i
}

__global__ void k_gcn_wT(const float* __restrict__ w, bhalf* __restrict__ wT, int K, int KP){
  int idx = blockIdx.x*blockDim.x + threadIdx.x;
  if (idx >= Dm*KP) return;
  int k = idx % KP, n = idx / KP;
  wT[idx] = (k < K) ? f2bf(w[k*Dm + n]) : (bhalf)0;   // (K,512) -> N-major, zero pad
}

__global__ void k_attP(const float* __restrict__ att, bhalf* __restrict__ aP, int Nr, int Npad){
  int idx = blockIdx.x*blockDim.x + threadIdx.x;
  if (idx >= Npad*64) return;
  int m = idx % 64, n = idx / 64;
  aP[idx] = (n < Nr && m < Nr) ? f2bf(att[n*Nr + m]) : (bhalf)0;
}

__global__ void k_scale_in(const float* __restrict__ src, bhalf* __restrict__ o, int tOff, int Tn){
  int idx = blockIdx.x*blockDim.x + threadIdx.x;
  if (idx >= Bn*Tn*Ff) return;
  int f = idx % Ff, t = (idx / Ff) % Tn, b = idx / (Ff*Tn);
  o[idx] = f2bf(src[(b*Tt + tOff + t)*Ff + f] * 0.001f);
}

__global__ void k_src_value(const float* __restrict__ src, const float* __restrict__ d, bhalf* __restrict__ sv){
  int idx = blockIdx.x*blockDim.x + threadIdx.x;
  if (idx >= Bn*Vn*Ff*Dct) return;
  int k = idx % Dct, f = (idx / Dct) % Ff, v = (idx / (Dct*Ff)) % Vn, b = idx / (Dct*Ff*Vn);
  float s = 0.f;
  for (int t=0;t<Vl;t++) s += d[k*Vl + t] * src[(b*Tt + v + t)*Ff + f];
  sv[idx] = f2bf(s);
}

__global__ void k_score(const bhalf* __restrict__ keyO, const bhalf* __restrict__ qO,
                        float* __restrict__ sc, float* __restrict__ scOut){
  int idx = blockIdx.x*blockDim.x + threadIdx.x;
  if (idx >= Bn*Vn) return;
  int v = idx % Vn, b = idx / Vn;
  float s = 0.f;
  for (int c=0;c<Dm;c++) s += bf2f(qO[b*Dm + c]) * bf2f(keyO[(b*Vn + v)*Dm + c]);
  s += 1e-15f;
  sc[idx] = s; scOut[idx] = s;
}

__global__ void k_dct_att(const float* __restrict__ sc, const bhalf* __restrict__ sv, float* __restrict__ da){
  int idx = blockIdx.x*blockDim.x + threadIdx.x;
  if (idx >= Bn*Ff*Dct) return;
  int k = idx % Dct, f = (idx / Dct) % Ff, b = idx / (Dct*Ff);
  float s = 0.f;
  for (int v=0;v<Vn;v++) s += sc[b*Vn + v] * bf2f(sv[((b*Vn + v)*Ff + f)*Dct + k]);
  da[idx] = s;
}

__global__ void k_out0(const float* __restrict__ d, const float* __restrict__ da, float* __restrict__ o){
  int idx = blockIdx.x*blockDim.x + threadIdx.x;
  if (idx >= Bn*Vl*Ff) return;
  int f = idx % Ff, t = (idx / Ff) % Vl, b = idx / (Ff*Vl);
  float s = 0.f;
  for (int k=0;k<Dct;k++) s += d[k*Vl + t] * da[(b*Ff + f)*Dct + k];   // idct = dct^T
  o[idx] = s;
}

// dct_in (concat of dct_initial | dct_att): f32 stride 20, bf16 stride 32 zero-padded
__global__ void k_dct_initial(const float* __restrict__ src, const float* __restrict__ d,
                              const float* __restrict__ da, float* __restrict__ dinF, bhalf* __restrict__ dinB){
  int idx = blockIdx.x*blockDim.x + threadIdx.x;
  if (idx >= Bn*Ff*32) return;
  int k = idx % 32, f = (idx / 32) % Ff, b = idx / (32*Ff);
  if (k >= 2*Dct) { dinB[idx] = (bhalf)0; return; }
  float s;
  if (k < Dct) {
    s = 0.f;
    for (int t=0;t<Vl;t++) {
      int g = (t < 10) ? (40 + t) : 49;    // gidx
      s += d[k*Vl + t] * src[(b*Tt + g)*Ff + f];
    }
  } else {
    s = da[(b*Ff + f)*Dct + (k - Dct)];
  }
  dinF[(b*Ff + f)*(2*Dct) + k] = s;
  dinB[idx] = f2bf(s);
}

__global__ void k_gc7_mm(const bhalf* __restrict__ y, const float* __restrict__ w7,
                         float* __restrict__ t7, int Npad, int Nr, int Fo){
  int idx = blockIdx.x*blockDim.x + threadIdx.x;
  if (idx >= Bn*Nr*Fo) return;
  int f = idx % Fo, n = (idx / Fo) % Nr, b = idx / (Fo*Nr);
  const bhalf* yr = y + (b*Npad + n)*Dm;
  float s = 0.f;
  for (int c=0;c<Dm;c++) s += bf2f(yr[c]) * w7[c*Fo + f];
  t7[idx] = s;
}

__global__ void k_gc7_att(const float* __restrict__ t7, const float* __restrict__ att7,
                          const float* __restrict__ b7, const float* __restrict__ residR,
                          int residT, float* __restrict__ o, int Nr, int Fo){
  int idx = blockIdx.x*blockDim.x + threadIdx.x;
  if (idx >= Bn*Nr*Fo) return;
  int f = idx % Fo, n = (idx / Fo) % Nr, b = idx / (Fo*Nr);
  float s = 0.f;
  for (int m=0;m<Nr;m++) s += att7[n*Nr + m] * t7[(b*Nr + m)*Fo + f];
  s += b7[f];
  s += residT ? residR[(b*Fo + f)*Nr + n] : residR[(b*Nr + n)*Fo + f];
  o[idx] = s;
}

// s1 (B,48,20) -> xt (B,32,64) bf16, transposed + zero padded rows/cols
__global__ void k_transpose_pad(const float* __restrict__ s1, bhalf* __restrict__ xt){
  int idx = blockIdx.x*blockDim.x + threadIdx.x;
  if (idx >= Bn*32*64) return;
  int c = idx % 64, kr = (idx / 64) % 32, b = idx / (64*32);
  xt[idx] = (kr < 2*Dct && c < Ff) ? f2bf(s1[(b*Ff + c)*(2*Dct) + kr]) : (bhalf)0;
}

__global__ void k_out_idct(const float* __restrict__ d, const float* __restrict__ t1, float* __restrict__ o){
  int idx = blockIdx.x*blockDim.x + threadIdx.x;
  if (idx >= Bn*Vl*Ff) return;
  int f = idx % Ff, t = (idx / Ff) % Vl, b = idx / (Ff*Vl);
  float s = 0.f;
  for (int k=0;k<Dct;k++) s += d[k*Vl + t] * t1[(b*2*Dct + k)*Ff + f];
  o[idx] = s;
}

__global__ void k_dct_next(const float* __restrict__ d, const float* __restrict__ o1,
                           float* __restrict__ dinF, bhalf* __restrict__ dinB){
  int idx = blockIdx.x*blockDim.x + threadIdx.x;
  if (idx >= Bn*Ff*Dct) return;
  int k = idx % Dct, f = (idx / Dct) % Ff, b = idx / (Dct*Ff);
  float s = 0.f;
  for (int t=0;t<Vl;t++) s += d[k*Vl + t] * o1[(b*Vl + t)*Ff + f];
  dinF[(b*Ff + f)*(2*Dct) + Dct + k] = s;
  dinB[(b*Ff + f)*32 + Dct + k]      = f2bf(s);
}

// ------------------- GCN core (templated host driver) -------------------
template<int NPAD, int NR, int FIN>
static const bhalf* run_core(hipStream_t stream, void* const* d_in, int P,
                             const bhalf* Xbf, const bhalf* w1T,
                             bhalf* const* bws, bhalf* const* aPs,
                             bhalf* tmpT, bhalf* yA, bhalf* yB, bhalf* zb)
{
  constexpr int KP  = (FIN  + 31) & ~31;
  constexpr int KKA = (NPAD + 31) & ~31;
  const int M = Bn * NPAD;
  auto L = [&](int o)->const float* { return (const float*)d_in[P + o]; };

  k_wmma_gemm<RM_DENSE,1,1,1,KP,KP,1,NPAD><<<dim3(2,M/64),dim3(256),0,stream>>>(
      Xbf, w1T, KP, tmpT, 0, nullptr,nullptr,nullptr,nullptr);
  k_wmma_gemm<RM_ATT,1,NPAD,1,64,64,2,1><<<dim3(2,M/64),dim3(256),0,stream>>>(
      aPs[0], tmpT, KKA, yA, NR, L(23), L(21), L(20), nullptr);
  const bhalf* yin = yA; bhalf* yout = yB;
  for (int bl=0; bl<2; bl++){
    const int Q = 10*bl;
    k_wmma_gemm<RM_DENSE,1,1,1,512,512,1,NPAD><<<dim3(2,M/64),dim3(256),0,stream>>>(
        yin, bws[2*bl], 512, tmpT, 0, nullptr,nullptr,nullptr,nullptr);
    k_wmma_gemm<RM_ATT,1,NPAD,1,64,64,2,1><<<dim3(2,M/64),dim3(256),0,stream>>>(
        aPs[1+2*bl], tmpT, KKA, zb, NR, L(Q+5), L(Q+1), L(Q+0), nullptr);
    k_wmma_gemm<RM_DENSE,1,1,1,512,512,1,NPAD><<<dim3(2,M/64),dim3(256),0,stream>>>(
        zb, bws[2*bl+1], 512, tmpT, 0, nullptr,nullptr,nullptr,nullptr);
    k_wmma_gemm<RM_ATT,1,NPAD,1,64,64,2,1><<<dim3(2,M/64),dim3(256),0,stream>>>(
        aPs[2+2*bl], tmpT, KKA, yout, NR, L(Q+8), L(Q+3), L(Q+2), yin);
    const bhalf* t = yin; yin = yout; yout = (bhalf*)t;
  }
  return yin;
}

// ------------------------- host orchestration -------------------------
#define LAUNCH(kern, total, ...) kern<<<dim3((unsigned)(((total)+255)/256)), dim3(256), 0, stream>>>(__VA_ARGS__)

extern "C" void kernel_launch(void* const* d_in, const int* in_sizes, int n_in,
                              void* d_out, int out_size, void* d_ws, size_t ws_size,
                              hipStream_t stream)
{
  (void)in_sizes; (void)n_in; (void)out_size; (void)ws_size;
  const float* src = (const float*)d_in[0];
  const float* wQ1 = (const float*)d_in[4];
  const float* wQ2 = (const float*)d_in[5];
  const float* wK1 = (const float*)d_in[6];
  const float* wK2 = (const float*)d_in[7];
  float* out    = (float*)d_out;
  float* out0p  = out;
  float* out1p  = out + Bn*Vl*Ff;
  float* out2p  = out + 2*Bn*Vl*Ff;
  float* scorep = out + 3*Bn*Vl*Ff;

  // GCN param leaves (jax sorted-key flattening, 28 leaves per gcn):
  // [0..9]  block0: bn1_b,bn1_g,bn2_b,bn2_g,gc1.att,gc1.b,gc1.w,gc2.att,gc2.b,gc2.w
  // [10..19]block1 ; [20]bn1_b [21]bn1_g [22]gc1.att [23]gc1.b [24]gc1.w
  // [25]gc7.att [26]gc7.b [27]gc7.w
  auto L = [&](int base, int o)->const float* { return (const float*)d_in[base + o]; };
  const int GS1 = 8, GT1 = 36, GS2 = 64, GT2 = 92;

  char* wsb = (char*)d_ws;
  size_t off = 0;
  auto alloc = [&](size_t bytes)->char* {
    char* p = wsb + off; off += (bytes + 255) & ~(size_t)255; return p;
  };

  // persistent buffers
  float* dct  = (float*)alloc(Dct*Vl*4);
  float* dinF = (float*)alloc((size_t)Bn*Ff*2*Dct*4);
  bhalf* dinB = (bhalf*)alloc((size_t)Bn*Ff*32*2);        // stride 32, zero padded
  bhalf* w1T  = (bhalf*)alloc((size_t)Dm*64*2);           // stride 32 or 64
  bhalf* bw0  = (bhalf*)alloc((size_t)Dm*Dm*2);
  bhalf* bw1  = (bhalf*)alloc((size_t)Dm*Dm*2);
  bhalf* bw2  = (bhalf*)alloc((size_t)Dm*Dm*2);
  bhalf* bw3  = (bhalf*)alloc((size_t)Dm*Dm*2);
  bhalf* aP0  = (bhalf*)alloc((size_t)48*64*2);
  bhalf* aP1  = (bhalf*)alloc((size_t)48*64*2);
  bhalf* aP2  = (bhalf*)alloc((size_t)48*64*2);
  bhalf* aP3  = (bhalf*)alloc((size_t)48*64*2);
  bhalf* aP4  = (bhalf*)alloc((size_t)48*64*2);
  float* t7   = (float*)alloc((size_t)Bn*48*20*4);
  float* s1   = (float*)alloc((size_t)Bn*48*20*4);
  bhalf* xt   = (bhalf*)alloc((size_t)Bn*32*64*2);        // stride 64, zero padded
  float* t1   = (float*)alloc((size_t)Bn*20*48*4);

  size_t offBig = off;
  // front-end scratch (dead before GCN phase; overlapped below)
  bhalf* Akey = (bhalf*)alloc((size_t)Bn*25*Ff*2);
  bhalf* Aq   = (bhalf*)alloc((size_t)Bn*10*Ff*2);
  bhalf* wK1T = (bhalf*)alloc((size_t)Dm*288*2);
  bhalf* wQ1T = (bhalf*)alloc((size_t)Dm*288*2);
  bhalf* wK2T = (bhalf*)alloc((size_t)Dm*2560*2);
  bhalf* wQ2T = (bhalf*)alloc((size_t)Dm*2560*2);
  bhalf* h1k  = (bhalf*)alloc((size_t)Bn*20*Dm*2);
  bhalf* h1q  = (bhalf*)alloc((size_t)Bn*5*Dm*2);
  bhalf* keyO = (bhalf*)alloc((size_t)Bn*16*Dm*2);
  bhalf* qO   = (bhalf*)alloc((size_t)Bn*Dm*2);
  bhalf* sval = (bhalf*)alloc((size_t)Bn*Vn*Ff*Dct*2);
  float* sc   = (float*)alloc((size_t)Bn*Vn*4);
  float* da   = (float*)alloc((size_t)Bn*Ff*Dct*4);

  // GCN big buffers reuse the same arena
  off = offBig;
  bhalf* tmpT = (bhalf*)alloc((size_t)Bn*Dm*64*2);        // 64-row stride, zero padded
  bhalf* yA   = (bhalf*)alloc((size_t)Bn*48*Dm*2);
  bhalf* yB   = (bhalf*)alloc((size_t)Bn*48*Dm*2);
  bhalf* zb   = (bhalf*)alloc((size_t)Bn*48*Dm*2);
  bhalf* bws[4] = {bw0,bw1,bw2,bw3};
  bhalf* aPs[5] = {aP0,aP1,aP2,aP3,aP4};

  // ---------------- front end ----------------
  LAUNCH(k_build_dct, Dct*Vl, dct);
  LAUNCH(k_conv_wT, Dm*48*6,  wK1, wK1T, 48, 6);
  LAUNCH(k_conv_wT, Dm*48*6,  wQ1, wQ1T, 48, 6);
  LAUNCH(k_conv_wT, Dm*512*5, wK2, wK2T, 512, 5);
  LAUNCH(k_conv_wT, Dm*512*5, wQ2, wQ2T, 512, 5);
  LAUNCH(k_scale_in, Bn*25*Ff, src, Akey, 0, 25);
  LAUNCH(k_scale_in, Bn*10*Ff, src, Aq, 40, 10);
  // key: len25 --k6--> 20 --k5--> 16 ; query: len10 -> 5 -> 1 (conv folded in K-loop)
  k_wmma_gemm<RM_CONV,25,20,48,48,288,0,1><<<dim3(2,(Bn*20)/64),dim3(256),0,stream>>>(
      Akey, wK1T, 288, h1k, 0, nullptr,nullptr,nullptr,nullptr);
  k_wmma_gemm<RM_CONV,20,16,512,512,2560,0,1><<<dim3(2,(Bn*16)/64),dim3(256),0,stream>>>(
      h1k, wK2T, 2560, keyO, 0, nullptr,nullptr,nullptr,nullptr);
  k_wmma_gemm<RM_CONV,10,5,48,48,288,0,1><<<dim3(2,(Bn*5)/64),dim3(256),0,stream>>>(
      Aq, wQ1T, 288, h1q, 0, nullptr,nullptr,nullptr,nullptr);
  k_wmma_gemm<RM_CONV,5,1,512,512,2560,0,1><<<dim3(2,(Bn*1)/64),dim3(256),0,stream>>>(
      h1q, wQ2T, 2560, qO, 0, nullptr,nullptr,nullptr,nullptr);
  LAUNCH(k_src_value, Bn*Vn*Ff*Dct, src, dct, sval);
  LAUNCH(k_score, Bn*Vn, keyO, qO, sc, scorep);
  LAUNCH(k_dct_att, Bn*Ff*Dct, sc, sval, da);
  LAUNCH(k_out0, Bn*Vl*Ff, dct, da, out0p);
  LAUNCH(k_dct_initial, Bn*Ff*32, src, dct, da, dinF, dinB);

  // ---------------- GCN phase ----------------
  LAUNCH(k_fill0, (int)((size_t)Bn*Dm*64/2), (u32t*)tmpT, (int)((size_t)Bn*Dm*64/2));

  auto prep_gcn = [&](int P, int Fin, int KP, int Nr, int Npad){
    LAUNCH(k_gcn_wT, Dm*KP, L(P,24), w1T, Fin, KP);
    LAUNCH(k_gcn_wT, Dm*Dm, L(P,6),  bw0, Dm, Dm);
    LAUNCH(k_gcn_wT, Dm*Dm, L(P,9),  bw1, Dm, Dm);
    LAUNCH(k_gcn_wT, Dm*Dm, L(P,16), bw2, Dm, Dm);
    LAUNCH(k_gcn_wT, Dm*Dm, L(P,19), bw3, Dm, Dm);
    LAUNCH(k_attP, Npad*64, L(P,22), aP0, Nr, Npad);
    LAUNCH(k_attP, Npad*64, L(P,4),  aP1, Nr, Npad);
    LAUNCH(k_attP, Npad*64, L(P,7),  aP2, Nr, Npad);
    LAUNCH(k_attP, Npad*64, L(P,14), aP3, Nr, Npad);
    LAUNCH(k_attP, Npad*64, L(P,17), aP4, Nr, Npad);
  };

  auto run_sgcn = [&](int P){
    prep_gcn(P, 2*Dct, 32, 48, 48);
    const bhalf* yf = run_core<48,48,20>(stream, d_in, P, dinB, w1T, bws, aPs, tmpT, yA, yB, zb);
    LAUNCH(k_gc7_mm,  Bn*48*20, yf, L(P,27), t7, 48, 48, 20);
    LAUNCH(k_gc7_att, Bn*48*20, t7, L(P,25), L(P,26), dinF, 0, s1, 48, 20);
  };
  auto run_tgcn = [&](int P){
    LAUNCH(k_transpose_pad, Bn*32*64, s1, xt);
    prep_gcn(P, 48, 64, 20, 32);
    const bhalf* yf = run_core<32,20,48>(stream, d_in, P, xt, w1T, bws, aPs, tmpT, yA, yB, zb);
    LAUNCH(k_gc7_mm,  Bn*20*48, yf, L(P,27), t7, 32, 20, 48);
    LAUNCH(k_gc7_att, Bn*20*48, t7, L(P,25), L(P,26), s1, 1, t1, 20, 48);
  };

  run_sgcn(GS1);
  run_tgcn(GT1);
  LAUNCH(k_out_idct, Bn*Vl*Ff, dct, t1, out1p);
  LAUNCH(k_dct_next, Bn*Ff*Dct, dct, out1p, dinF, dinB);
  run_sgcn(GS2);
  run_tgcn(GT2);
  LAUNCH(k_out_idct, Bn*Vl*Ff, dct, t1, out2p);
}
